// CausalSelfAttention_12687333392718
// MI455X (gfx1250) — matrix operand order repaired and checked by reading, not compiled, not run on hardware
//
#include <hip/hip_runtime.h>
#include <hip/hip_bf16.h>
#include <stdint.h>

// ---------------------------------------------------------------------------
// Causal self-attention for MI455X (gfx1250, wave32, WMMA).
//  B=2, T=4096, C=768, H=12, HD=64
// Pipeline: f32->f16 convert -> WMMA GEMM (qkv) -> flash-attention (WMMA +
// double-buffered TDM tensor_load_to_lds for K/V tiles) -> WMMA GEMM (proj).
// ---------------------------------------------------------------------------

typedef __attribute__((ext_vector_type(16))) _Float16 v16h;
typedef __attribute__((ext_vector_type(8)))  _Float16 v8h;
typedef __attribute__((ext_vector_type(8)))  float    v8f;
typedef __attribute__((ext_vector_type(4)))  unsigned int u32x4;
typedef __attribute__((ext_vector_type(8)))  unsigned int u32x8;

#define BB  2
#define TT  4096
#define CC  768
#define HH  12
#define HD  64
#define QS  (3 * CC)      // 2304, qkv row stride
#define MM  (BB * TT)     // 8192

union AFrag { v16h v; v8h h[2]; };

__device__ __forceinline__ v8f wmma_f16(v16h a, v16h b, v8f c) {
    return __builtin_amdgcn_wmma_f32_16x16x32_f16(false, a, false, b,
                                                  (short)0, c, false, false);
}

// A-fragment (16x32 f16, MxK): lanes 0-15 hold K {0..7,16..23}, lanes 16-31
// hold K {8..15,24..31}; row = lane % 16.  src = row base (K-contiguous).
__device__ __forceinline__ v16h load_afrag(const _Float16* src, int hf) {
    AFrag a;
    a.h[0] = *(const v8h*)(src + hf * 8);
    a.h[1] = *(const v8h*)(src + 16 + hf * 8);
    return a.v;
}

// B-fragment (32x16 f16, KxN): lanes 0-15 hold K {0..7,16..23}, lanes 16-31
// hold K {8..15,24..31}; col = lane % 16.  src = column base laid out K-contiguous.
__device__ __forceinline__ v16h load_bfrag(const _Float16* src, int hf) {
    AFrag b;
    b.h[0] = *(const v8h*)(src + hf * 8);
    b.h[1] = *(const v8h*)(src + 16 + hf * 8);
    return b.v;
}

// ---------------------------------------------------------------------------
// elementwise f32 -> f16
// ---------------------------------------------------------------------------
__global__ void cvt_f32_f16(const float* __restrict__ s,
                            _Float16* __restrict__ d, int n) {
    int i = blockIdx.x * blockDim.x + threadIdx.x;
    if (i < n) d[i] = (_Float16)s[i];
}

// ---------------------------------------------------------------------------
// GEMM: out[m,n] = sum_k A[m*K+k] * W[n*K+k]   (both K-contiguous)
// block = 128 threads = 4 waves; wave -> 32x64 tile (2 A x 4 B frags,
// 8 wmma per k-step => 0.75 fragment-loads per wmma); block -> 128x64.
// ---------------------------------------------------------------------------
template <bool OUT_F16>
__global__ __launch_bounds__(128) void gemm_wmma(const _Float16* __restrict__ A,
                                                 const _Float16* __restrict__ W,
                                                 void* __restrict__ outp,
                                                 int N, int K) {
    const int tid  = threadIdx.x;
    const int wave = tid >> 5;
    const int lane = tid & 31;
    const int hf   = lane >> 4;
    const int lr   = lane & 15;

    const int m0 = blockIdx.y * 128 + wave * 32;
    const int n0 = blockIdx.x * 64;

    const _Float16* arow0 = A + (size_t)(m0 + lr) * K;
    const _Float16* arow1 = A + (size_t)(m0 + 16 + lr) * K;
    const _Float16* wr0   = W + (size_t)(n0 + lr) * K;

    v8f acc[8];                       // [row-half][ncol] : acc[j], acc[4+j]
#pragma unroll
    for (int j = 0; j < 8; ++j) acc[j] = (v8f)0.0f;

    for (int k0 = 0; k0 < K; k0 += 32) {
        if (k0 + 32 < K) {
            __builtin_prefetch(arow0 + k0 + 32, 0, 0);
            __builtin_prefetch(arow1 + k0 + 32, 0, 0);
            __builtin_prefetch(wr0 + k0 + 32, 0, 0);
        }
        v16h a0 = load_afrag(arow0 + k0, hf);
        v16h a1 = load_afrag(arow1 + k0, hf);
#pragma unroll
        for (int j = 0; j < 4; ++j) {
            v16h bf = load_bfrag(wr0 + (size_t)j * 16 * K + k0, hf);
            acc[j]     = wmma_f16(a0, bf, acc[j]);
            acc[4 + j] = wmma_f16(a1, bf, acc[4 + j]);
        }
    }

    // C layout: VGPR r -> row hf*8+r, col = lane%16
#pragma unroll
    for (int i = 0; i < 2; ++i) {
#pragma unroll
        for (int j = 0; j < 4; ++j) {
#pragma unroll
            for (int r = 0; r < 8; ++r) {
                size_t idx = (size_t)(m0 + i * 16 + hf * 8 + r) * N
                           + n0 + j * 16 + lr;
                float v = acc[i * 4 + j][r];
                if (OUT_F16) ((_Float16*)outp)[idx] = (_Float16)v;
                else         ((float*)outp)[idx]    = v;
            }
        }
    }
}

// ---------------------------------------------------------------------------
// TDM: async-load a 32x64 f16 tile (row stride QS elements) into LDS.
// D# per CDNA5 ISA ch.10.7/§8; 2-SGPR-group form (2D tensor).
// ---------------------------------------------------------------------------
__device__ __forceinline__ void tdm_load_tile(unsigned int lds_byte_off,
                                              const _Float16* gsrc) {
    unsigned long long ga = (unsigned long long)(uintptr_t)gsrc;
    u32x4 g0;
    g0.x = 1u;                                   // count=1, user descriptor
    g0.y = lds_byte_off;                         // lds_addr
    g0.z = (unsigned int)(ga & 0xFFFFFFFFull);   // global_addr[31:0]
    g0.w = (unsigned int)((ga >> 32) & 0x01FFFFFFull) | (2u << 30); // [56:32]|type=2
    u32x8 g1;
    g1.s0 = (1u << 16);                          // workgroup_mask=0, data_size=2B
    g1.s1 = ((unsigned)HD & 0xFFFFu) << 16;      // tensor_dim0 lo16 (=64)
    g1.s2 = ((unsigned)TT & 0xFFFFu) << 16;      // dim0 hi16(0) | tensor_dim1 lo16
    g1.s3 = (((unsigned)TT >> 16) & 0xFFFFu) | ((unsigned)HD << 16); // dim1 hi|tile_dim0
    g1.s4 = 32u;                                 // tile_dim1=32, tile_dim2=0
    g1.s5 = (unsigned)QS;                        // tensor_dim0_stride lo32
    g1.s6 = 0u;
    g1.s7 = 0u;
    asm volatile("tensor_load_to_lds %0, %1" :: "s"(g0), "s"(g1) : "memory");
}

// ---------------------------------------------------------------------------
// Flash attention: one workgroup per (b, h, 64-row q block); 4 waves,
// each wave owns a 16-row q strip.  Keys processed 32 at a time with
// double-buffered TDM K/V tiles (DMA of tile i+1 overlaps compute on tile i).
// ---------------------------------------------------------------------------
__global__ __launch_bounds__(128) void attn_wmma(const _Float16* __restrict__ qkv,
                                                 _Float16* __restrict__ outh) {
    __shared__ __align__(16) _Float16 kT[2][32 * HD];   // K tiles [t][d]
    __shared__ __align__(16) _Float16 vT[2][32 * HD];   // V tiles [t][d]
    __shared__ __align__(16) _Float16 vtT[HD * 32];     // V tile  [d][t]
    __shared__ __align__(16) _Float16 pT[4][16 * 32];   // per-wave P staging

    const int tid  = threadIdx.x;
    const int wave = tid >> 5;
    const int lane = tid & 31;
    const int hf   = lane >> 4;
    const int lr   = lane & 15;

    const int b  = blockIdx.z;
    const int h  = blockIdx.y;
    const int qb = blockIdx.x * 64;

    const unsigned kOff0 = (unsigned)(uintptr_t)(void*)&kT[0][0];
    const unsigned kOff1 = (unsigned)(uintptr_t)(void*)&kT[1][0];
    const unsigned vOff0 = (unsigned)(uintptr_t)(void*)&vT[0][0];
    const unsigned vOff1 = (unsigned)(uintptr_t)(void*)&vT[1][0];

    // Q fragments for this wave's 16 rows (loaded once; K dim = 64 -> 2 frags)
    const _Float16* qrow =
        qkv + (size_t)(b * TT + qb + wave * 16 + lr) * QS + h * HD;
    v16h qa0 = load_afrag(qrow +  0, hf);
    v16h qa1 = load_afrag(qrow + 32, hf);

    v8f acc[4];
#pragma unroll
    for (int j = 0; j < 4; ++j) acc[j] = (v8f)0.0f;
    float mrow[8], lrow[8];
#pragma unroll
    for (int r = 0; r < 8; ++r) { mrow[r] = -1e30f; lrow[r] = 0.0f; }

    const _Float16* kvbase = qkv + (size_t)b * TT * QS + h * HD;
    const int nIter = (qb + 64) / 32;             // causal bound / key tile

    // prologue: fetch tile 0 into buffer 0 (wave 0, scalar-guarded)
    if (__builtin_amdgcn_readfirstlane(tid) == 0) {
        tdm_load_tile(kOff0, kvbase + CC);
        tdm_load_tile(vOff0, kvbase + 2 * CC);
    }

    for (int it = 0; it < nIter; ++it) {
        const int t0  = it * 32;
        const int buf = it & 1;
        const bool more = (it + 1) < nIter;

        // issue TDM for next tile into the other buffer, then wait for
        // the current tile only (TDM completes in-order per wave).
        if (__builtin_amdgcn_readfirstlane(tid) == 0) {
            if (more) {
                const _Float16* nb = kvbase + (size_t)(t0 + 32) * QS;
                tdm_load_tile(buf ? kOff0 : kOff1, nb + CC);
                tdm_load_tile(buf ? vOff0 : vOff1, nb + 2 * CC);
            }
        }
        if (more) __builtin_amdgcn_s_wait_tensorcnt(2);
        else      __builtin_amdgcn_s_wait_tensorcnt(0);
        __syncthreads();                      // current K/V tile visible

        // transpose V -> vtT[d][t] so PV B-fragments are contiguous
        for (int i = tid; i < 32 * HD; i += 128) {
            int t = i >> 6, d = i & 63;
            vtT[d * 32 + t] = vT[buf][i];
        }
        __syncthreads();

        // --- S = Q K^T  (two 16x16 key sub-tiles, K dim 64 = 2 wmma each) ---
        v8f s0 = (v8f)0.0f, s1 = (v8f)0.0f;
        {
            const _Float16* kb0 = &kT[buf][0] + (size_t)(0 * 16 + lr) * HD;
            const _Float16* kb1 = &kT[buf][0] + (size_t)(1 * 16 + lr) * HD;
            s0 = wmma_f16(qa0, load_bfrag(kb0 +  0, hf), s0);
            s0 = wmma_f16(qa1, load_bfrag(kb0 + 32, hf), s0);
            s1 = wmma_f16(qa0, load_bfrag(kb1 +  0, hf), s1);
            s1 = wmma_f16(qa1, load_bfrag(kb1 + 32, hf), s1);
        }

        // --- scale + causal mask + online softmax ---
        const int qg = qb + wave * 16 + hf * 8;   // + r = global q row
#pragma unroll
        for (int r = 0; r < 8; ++r) {
            float v0 = s0[r] * 0.125f;            // 1/sqrt(64)
            float v1 = s1[r] * 0.125f;
            if (t0 + lr      > qg + r) v0 = -1e30f;
            if (t0 + 16 + lr > qg + r) v1 = -1e30f;

            float rm = fmaxf(v0, v1);             // row max across 16 lanes
#pragma unroll
            for (int mk = 1; mk < 16; mk <<= 1)
                rm = fmaxf(rm, __shfl_xor(rm, mk, 32));

            float mnew = fmaxf(mrow[r], rm);
            float sc   = __expf(mrow[r] - mnew);
            v0 = __expf(v0 - mnew);
            v1 = __expf(v1 - mnew);

            float rs = v0 + v1;                   // row sum across 16 lanes
#pragma unroll
            for (int mk = 1; mk < 16; mk <<= 1)
                rs += __shfl_xor(rs, mk, 32);

            lrow[r] = lrow[r] * sc + rs;
            mrow[r] = mnew;
#pragma unroll
            for (int j = 0; j < 4; ++j) acc[j][r] *= sc;
            s0[r] = v0;
            s1[r] = v1;
        }

        // --- stage P (C layout -> A-fragment layout) through this wave's LDS ---
        _Float16* pw = pT[wave];
#pragma unroll
        for (int r = 0; r < 8; ++r) {
            pw[(hf * 8 + r) * 32 +      lr] = (_Float16)s0[r];
            pw[(hf * 8 + r) * 32 + 16 + lr] = (_Float16)s1[r];
        }
        v16h pa = load_afrag(pw + lr * 32, hf);   // per-wave, DS ops in-order

        // --- acc += P (16x32) x V (32x64) ---
#pragma unroll
        for (int j = 0; j < 4; ++j) {
            const _Float16* vb = vtT + (size_t)(j * 16 + lr) * 32;
            acc[j] = wmma_f16(pa, load_bfrag(vb, hf), acc[j]);
        }
        __syncthreads();   // all waves done with vtT and the buffer that the
                           // iteration after next will overwrite via TDM
    }

    // --- normalize and store merged-head f16 output ---
#pragma unroll
    for (int j = 0; j < 4; ++j) {
#pragma unroll
        for (int r = 0; r < 8; ++r) {
            int row = qb + wave * 16 + hf * 8 + r;
            outh[(size_t)(b * TT + row) * CC + h * HD + j * 16 + lr] =
                (_Float16)(acc[j][r] / lrow[r]);
        }
    }
}

// ---------------------------------------------------------------------------
// launcher
// ---------------------------------------------------------------------------
extern "C" void kernel_launch(void* const* d_in, const int* in_sizes, int n_in,
                              void* d_out, int out_size, void* d_ws, size_t ws_size,
                              hipStream_t stream) {
    const float* x  = (const float*)d_in[0];   // [B,T,C]
    const float* wa = (const float*)d_in[1];   // [3C,C]
    const float* wp = (const float*)d_in[2];   // [C,C]
    float* out = (float*)d_out;                // [B,T,C]

    char* ws = (char*)d_ws;
    const size_t nx  = (size_t)MM * CC;        // 6,291,456
    const size_t nwa = (size_t)QS * CC;        // 1,769,472
    const size_t nwp = (size_t)CC * CC;        //   589,824
    const size_t nqk = (size_t)MM * QS;        // 18,874,368

    _Float16* xh   = (_Float16*)(ws);
    _Float16* wah  = (_Float16*)(ws + nx * 2);
    _Float16* wph  = (_Float16*)(ws + (nx + nwa) * 2);
    _Float16* qkvh = (_Float16*)(ws + (nx + nwa + nwp) * 2);
    _Float16* atth = (_Float16*)(ws + (nx + nwa + nwp + nqk) * 2);

    cvt_f32_f16<<<dim3((unsigned)((nx  + 255) / 256)), 256, 0, stream>>>(x,  xh,  (int)nx);
    cvt_f32_f16<<<dim3((unsigned)((nwa + 255) / 256)), 256, 0, stream>>>(wa, wah, (int)nwa);
    cvt_f32_f16<<<dim3((unsigned)((nwp + 255) / 256)), 256, 0, stream>>>(wp, wph, (int)nwp);

    // qkv = x @ w_attn^T : [8192 x 2304], K=768
    gemm_wmma<true><<<dim3(QS / 64, MM / 128), 128, 0, stream>>>(xh, wah, qkvh, QS, CC);

    // flash attention -> atth [B,T,C] f16
    attn_wmma<<<dim3(TT / 64, HH, BB), 128, 0, stream>>>(qkvh, atth);

    // out = atth @ w_proj^T : [8192 x 768], K=768, f32 output
    gemm_wmma<false><<<dim3(CC / 64, MM / 128), 128, 0, stream>>>(atth, wph, out, CC, CC);
}
